// InternVisionLayer_14448269984354
// MI455X (gfx1250) — matrix operand order repaired and checked by reading, not compiled, not run on hardware
//
#include <hip/hip_runtime.h>
#include <math.h>

typedef __bf16 bf16;
typedef __attribute__((ext_vector_type(8)))  __bf16 bf16x8;
typedef __attribute__((ext_vector_type(16))) __bf16 bf16x16;
typedef __attribute__((ext_vector_type(8)))  float  f32x8;

union FragU { bf16x16 v; bf16x8 h[2]; };

__device__ __forceinline__ int imin(int a, int b) { return a < b ? a : b; }

// ---- WMMA wrapper: D = A(16x32 bf16) * B(32x16 bf16) + C(16x16 f32) ----
__device__ __forceinline__ f32x8 wmma_bf16(bf16x16 a, bf16x16 b, f32x8 c) {
  return __builtin_amdgcn_wmma_f32_16x16x32_bf16(
      /*neg_a=*/false, a, /*neg_b=*/false, b,
      /*c_mod=*/(short)0, c, /*reuse_a=*/false, /*reuse_b=*/false);
}

// A fragment 16x32: lane&15 = row; lane>=16 selects K halves {8..15,24..31}
__device__ __forceinline__ bf16x16 load_frag_a(const bf16* __restrict__ p, int ld,
                                               int row0, int k0) {
  int lane = threadIdx.x & 31;
  int r = lane & 15, hi = lane >> 4;
  const bf16* base = p + (size_t)(row0 + r) * ld + k0;
  FragU u;
  u.h[0] = *(const bf16x8*)(base + hi * 8);
  u.h[1] = *(const bf16x8*)(base + 16 + hi * 8);
  return u.v;
}
__device__ __forceinline__ bf16x16 load_frag_a_clamp(const bf16* __restrict__ p, int ld,
                                                     int row0, int rowMax, int k0) {
  int lane = threadIdx.x & 31;
  int r = lane & 15, hi = lane >> 4;
  const bf16* base = p + (size_t)imin(row0 + r, rowMax) * ld + k0;
  FragU u;
  u.h[0] = *(const bf16x8*)(base + hi * 8);
  u.h[1] = *(const bf16x8*)(base + 16 + hi * 8);
  return u.v;
}
// B fragment 32x16 for C = A * Mrows^T : lane&15 = column = row of M[N,K];
// K split: lanes<16 -> K {0..7,16..23}, lanes>=16 -> K {8..15,24..31}
__device__ __forceinline__ bf16x16 load_frag_b(const bf16* __restrict__ p, int ld,
                                               int row0, int k0) {
  int lane = threadIdx.x & 31;
  int n = lane & 15, hi = lane >> 4;
  const bf16* base = p + (size_t)(row0 + n) * ld + k0;
  FragU u;
  u.h[0] = *(const bf16x8*)(base + hi * 8);
  u.h[1] = *(const bf16x8*)(base + 16 + hi * 8);
  return u.v;
}
__device__ __forceinline__ bf16x16 load_frag_b_clamp(const bf16* __restrict__ p, int ld,
                                                     int row0, int rowMax, int k0) {
  int lane = threadIdx.x & 31;
  int n = lane & 15, hi = lane >> 4;
  const bf16* base = p + (size_t)imin(row0 + n, rowMax) * ld + k0;
  FragU u;
  u.h[0] = *(const bf16x8*)(base + hi * 8);
  u.h[1] = *(const bf16x8*)(base + 16 + hi * 8);
  return u.v;
}

// ---------------- f32 -> bf16 convert ----------------
__global__ __launch_bounds__(256) void cvt_kernel(const float* __restrict__ x,
                                                  bf16* __restrict__ y, int n) {
  for (int i = blockIdx.x * 256 + threadIdx.x; i < n; i += gridDim.x * 256)
    y[i] = (bf16)x[i];
}

// ---------------- LayerNorm (one row per block), output bf16 ----------------
__global__ __launch_bounds__(256) void ln_kernel(const float* __restrict__ x,
                                                 const float* __restrict__ g,
                                                 const float* __restrict__ bb,
                                                 bf16* __restrict__ y) {
  const int H = 1024;
  __shared__ float red[8];
  int row = blockIdx.x, tid = threadIdx.x;
  const float* xr = x + (size_t)row * H;
  float v[4], s = 0.f;
#pragma unroll
  for (int i = 0; i < 4; ++i) { v[i] = xr[tid + i * 256]; s += v[i]; }
#pragma unroll
  for (int o = 16; o; o >>= 1) s += __shfl_xor(s, o, 32);
  if ((tid & 31) == 0) red[tid >> 5] = s;
  __syncthreads();
  float tot = 0.f;
#pragma unroll
  for (int i = 0; i < 8; ++i) tot += red[i];
  float mean = tot * (1.f / 1024.f);
  __syncthreads();
  s = 0.f;
#pragma unroll
  for (int i = 0; i < 4; ++i) { float d = v[i] - mean; s += d * d; }
#pragma unroll
  for (int o = 16; o; o >>= 1) s += __shfl_xor(s, o, 32);
  if ((tid & 31) == 0) red[tid >> 5] = s;
  __syncthreads();
  tot = 0.f;
#pragma unroll
  for (int i = 0; i < 8; ++i) tot += red[i];
  float inv = rsqrtf(tot * (1.f / 1024.f) + 1e-5f);
#pragma unroll
  for (int i = 0; i < 4; ++i) {
    int c = tid + i * 256;
    y[(size_t)row * H + c] = (bf16)((v[i] - mean) * inv * g[c] + bb[c]);
  }
}

// ---------------- GEMM: C[T,N] = Abf[T,K] @ Wbf[N,K]^T + bias, epilogues ----
// Block tile 128x128, 8 waves, wave tile 32x64, two-stage K pipeline (step 64).
// EPI 0: store bf16     EPI 1: outf = res + ls*(acc+bias)   EPI 2: GELU -> bf16
template <int EPI>
__global__ __launch_bounds__(256) void gemm_kernel(
    const bf16* __restrict__ A, const bf16* __restrict__ W,
    const float* __restrict__ bias, const float* __restrict__ res,
    const float* __restrict__ ls, bf16* __restrict__ outb,
    float* __restrict__ outf, int T, int K, int N) {
  int nBlk = blockIdx.x * 128, mBlk = blockIdx.y * 128;
  int wid = threadIdx.x >> 5;
  int mOff = mBlk + (wid & 3) * 32;
  int nOff = nBlk + (wid >> 2) * 64;
  f32x8 acc[2][4] = {};

  // stage-even fragments at k=0
  bf16x16 a0 = load_frag_a_clamp(A, K, mOff, T - 1, 0);
  bf16x16 a1 = load_frag_a_clamp(A, K, mOff + 16, T - 1, 0);
  bf16x16 b0 = load_frag_b(W, K, nOff, 0);
  bf16x16 b1 = load_frag_b(W, K, nOff + 16, 0);
  bf16x16 b2 = load_frag_b(W, K, nOff + 32, 0);
  bf16x16 b3 = load_frag_b(W, K, nOff + 48, 0);

  for (int k = 0; k < K; k += 64) {
    // prefetch stage-odd at k+32 (always in range: K % 64 == 0)
    bf16x16 c0 = load_frag_a_clamp(A, K, mOff, T - 1, k + 32);
    bf16x16 c1 = load_frag_a_clamp(A, K, mOff + 16, T - 1, k + 32);
    bf16x16 d0 = load_frag_b(W, K, nOff, k + 32);
    bf16x16 d1 = load_frag_b(W, K, nOff + 16, k + 32);
    bf16x16 d2 = load_frag_b(W, K, nOff + 32, k + 32);
    bf16x16 d3 = load_frag_b(W, K, nOff + 48, k + 32);
    // compute stage-even
    acc[0][0] = wmma_bf16(a0, b0, acc[0][0]);
    acc[0][1] = wmma_bf16(a0, b1, acc[0][1]);
    acc[0][2] = wmma_bf16(a0, b2, acc[0][2]);
    acc[0][3] = wmma_bf16(a0, b3, acc[0][3]);
    acc[1][0] = wmma_bf16(a1, b0, acc[1][0]);
    acc[1][1] = wmma_bf16(a1, b1, acc[1][1]);
    acc[1][2] = wmma_bf16(a1, b2, acc[1][2]);
    acc[1][3] = wmma_bf16(a1, b3, acc[1][3]);
    // prefetch stage-even at k+64
    if (k + 64 < K) {
      a0 = load_frag_a_clamp(A, K, mOff, T - 1, k + 64);
      a1 = load_frag_a_clamp(A, K, mOff + 16, T - 1, k + 64);
      b0 = load_frag_b(W, K, nOff, k + 64);
      b1 = load_frag_b(W, K, nOff + 16, k + 64);
      b2 = load_frag_b(W, K, nOff + 32, k + 64);
      b3 = load_frag_b(W, K, nOff + 48, k + 64);
    }
    // compute stage-odd
    acc[0][0] = wmma_bf16(c0, d0, acc[0][0]);
    acc[0][1] = wmma_bf16(c0, d1, acc[0][1]);
    acc[0][2] = wmma_bf16(c0, d2, acc[0][2]);
    acc[0][3] = wmma_bf16(c0, d3, acc[0][3]);
    acc[1][0] = wmma_bf16(c1, d0, acc[1][0]);
    acc[1][1] = wmma_bf16(c1, d1, acc[1][1]);
    acc[1][2] = wmma_bf16(c1, d2, acc[1][2]);
    acc[1][3] = wmma_bf16(c1, d3, acc[1][3]);
  }

  int lane = threadIdx.x & 31;
  int cn = lane & 15, hi = lane >> 4;
#pragma unroll
  for (int tm = 0; tm < 2; ++tm)
#pragma unroll
    for (int tn = 0; tn < 4; ++tn) {
      int col = nOff + tn * 16 + cn;
      float bv = bias[col];
#pragma unroll
      for (int r = 0; r < 8; ++r) {
        int row = mOff + tm * 16 + r + hi * 8;
        if (row >= T) continue;
        float v = acc[tm][tn][r] + bv;
        size_t idx = (size_t)row * N + col;
        if (EPI == 0) {
          outb[idx] = (bf16)v;
        } else if (EPI == 1) {
          outf[idx] = res[idx] + ls[col] * v;
        } else {
          outb[idx] = (bf16)(0.5f * v * (1.f + erff(v * 0.70710678f)));
        }
      }
    }
}

// ---------------- Flash-style attention ----------------
// qkv: bf16 [B*S, 3072]; cols: q = h*64+d, k = 1024+h*64+d, v = 2048+h*64+d
// out: bf16 [B*S, 1024] (already [B,S,NH,HD] merged)
__global__ __launch_bounds__(128) void attn_kernel(const bf16* __restrict__ qkv,
                                                   bf16* __restrict__ out) {
  const int S = 1025, LD = 3072, NH = 16;
  __shared__ bf16 vt[64 * 32];        // V tile transposed: vt[d][kk]
  __shared__ bf16 pbuf[4][16 * 32];   // per-wave P tile [qrow][key]
  int b = blockIdx.x / NH, h = blockIdx.x % NH;
  int wid = threadIdx.x >> 5, lane = threadIdx.x & 31;
  int hi = lane >> 4;
  int qs0 = blockIdx.y * 64 + wid * 16;
  int rowMax = b * S + S - 1;
  const bf16* qp = qkv + h * 64;
  const bf16* kp = qkv + 1024 + h * 64;
  const bf16* vp = qkv + 2048 + h * 64;

  bf16x16 qf0 = load_frag_a_clamp(qp, LD, b * S + qs0, rowMax, 0);
  bf16x16 qf1 = load_frag_a_clamp(qp, LD, b * S + qs0, rowMax, 32);

  f32x8 o0 = {}, o1 = {}, o2 = {}, o3 = {};
  float mrow[8], lrow[8];
#pragma unroll
  for (int i = 0; i < 8; ++i) { mrow[i] = -3.0e38f; lrow[i] = 0.f; }

  const float scale = 0.125f;  // 1/sqrt(64)
  const int nTiles = (S + 31) / 32;  // 33
  for (int j = 0; j < nTiles; ++j) {
    int kBase = j * 32;
    // stage V tile transposed into LDS (all 4 waves cooperate)
    for (int t = threadIdx.x; t < 64 * 32; t += 128) {
      int kk = t & 31, d = t >> 5;
      int srow = imin(kBase + kk, S - 1);
      vt[d * 32 + kk] = vp[(size_t)(b * S + srow) * LD + d];
    }
    __syncthreads();

    // scores: two 16x16 C tiles over 32 keys (batch all K-fragment loads)
    bf16x16 kf0a = load_frag_b_clamp(kp, LD, b * S + kBase, rowMax, 0);
    bf16x16 kf0b = load_frag_b_clamp(kp, LD, b * S + kBase, rowMax, 32);
    bf16x16 kf1a = load_frag_b_clamp(kp, LD, b * S + kBase + 16, rowMax, 0);
    bf16x16 kf1b = load_frag_b_clamp(kp, LD, b * S + kBase + 16, rowMax, 32);
    f32x8 sc[2];
    {
      f32x8 t0 = {}, t1 = {};
      t0 = wmma_bf16(qf0, kf0a, t0);
      t1 = wmma_bf16(qf0, kf1a, t1);
      t0 = wmma_bf16(qf1, kf0b, t0);
      t1 = wmma_bf16(qf1, kf1b, t1);
      sc[0] = t0; sc[1] = t1;
    }
#pragma unroll
    for (int t = 0; t < 2; ++t) {
      bool ok = (kBase + t * 16 + (lane & 15)) < S;
#pragma unroll
      for (int q = 0; q < 8; ++q) sc[t][q] = ok ? sc[t][q] * scale : -3.0e38f;
    }
    // online softmax per row (rows live across 16 lanes of each half-wave)
#pragma unroll
    for (int q = 0; q < 8; ++q) {
      float mx = fmaxf(sc[0][q], sc[1][q]);
#pragma unroll
      for (int o = 8; o; o >>= 1) mx = fmaxf(mx, __shfl_xor(mx, o, 32));
      float mnew = fmaxf(mrow[q], mx);
      float alpha = __expf(mrow[q] - mnew);
      float p0 = __expf(sc[0][q] - mnew);
      float p1 = __expf(sc[1][q] - mnew);
      sc[0][q] = p0; sc[1][q] = p1;
      float ps = p0 + p1;
#pragma unroll
      for (int o = 8; o; o >>= 1) ps += __shfl_xor(ps, o, 32);
      lrow[q] = lrow[q] * alpha + ps;
      mrow[q] = mnew;
      o0[q] *= alpha; o1[q] *= alpha; o2[q] *= alpha; o3[q] *= alpha;
    }
    // P (C layout, f32) -> bf16 A layout via per-wave LDS buffer
    bf16* pb = pbuf[wid];
#pragma unroll
    for (int t = 0; t < 2; ++t)
#pragma unroll
      for (int q = 0; q < 8; ++q)
        pb[(q + 8 * hi) * 32 + t * 16 + (lane & 15)] = (bf16)sc[t][q];
    asm volatile("s_wait_dscnt 0" ::: "memory");
    bf16x16 pf = load_frag_a(pb, 32, 0, 0);
    bf16x16 vf0 = load_frag_b(vt, 32, 0, 0);
    bf16x16 vf1 = load_frag_b(vt, 32, 16, 0);
    bf16x16 vf2 = load_frag_b(vt, 32, 32, 0);
    bf16x16 vf3 = load_frag_b(vt, 32, 48, 0);
    o0 = wmma_bf16(pf, vf0, o0);
    o1 = wmma_bf16(pf, vf1, o1);
    o2 = wmma_bf16(pf, vf2, o2);
    o3 = wmma_bf16(pf, vf3, o3);
    __syncthreads();
  }
  // store O / l
#pragma unroll
  for (int q = 0; q < 8; ++q) {
    int s = qs0 + q + 8 * hi;
    if (s >= S) continue;
    float rinv = 1.f / lrow[q];
    size_t base = (size_t)(b * S + s) * 1024 + h * 64 + (lane & 15);
    out[base + 0]  = (bf16)(o0[q] * rinv);
    out[base + 16] = (bf16)(o1[q] * rinv);
    out[base + 32] = (bf16)(o2[q] * rinv);
    out[base + 48] = (bf16)(o3[q] * rinv);
  }
}

// ---------------- launch ----------------
extern "C" void kernel_launch(void* const* d_in, const int* in_sizes, int n_in,
                              void* d_out, int out_size, void* d_ws, size_t ws_size,
                              hipStream_t stream) {
  (void)in_sizes; (void)n_in; (void)out_size; (void)ws_size;
  const int T = 16 * 1025;  // 16400 tokens

  const float* hidden = (const float*)d_in[0];
  const float* n1g = (const float*)d_in[1];
  const float* n1b = (const float*)d_in[2];
  const float* qkv_w = (const float*)d_in[3];
  const float* qkv_b = (const float*)d_in[4];
  const float* proj_w = (const float*)d_in[5];
  const float* proj_b = (const float*)d_in[6];
  const float* ls1 = (const float*)d_in[7];
  const float* n2g = (const float*)d_in[8];
  const float* n2b = (const float*)d_in[9];
  const float* fc1_w = (const float*)d_in[10];
  const float* fc1_b = (const float*)d_in[11];
  const float* fc2_w = (const float*)d_in[12];
  const float* fc2_b = (const float*)d_in[13];
  const float* ls2 = (const float*)d_in[14];

  char* ws = (char*)d_ws;
  // offsets (bytes), all 256-aligned
  bf16* WQKV = (bf16*)(ws + 0);                   //  6,291,456
  bf16* WPROJ = (bf16*)(ws + 6291456);            //  2,097,152
  bf16* WFC1 = (bf16*)(ws + 8388608);             //  8,388,608
  bf16* WFC2 = (bf16*)(ws + 16777216);            //  8,388,608
  bf16* XBF  = (bf16*)(ws + 25165824);            // 33,587,200 (LN1 out, reused for LN2 out)
  bf16* QKVB = (bf16*)(ws + 58753024);            // 100,761,600
  bf16* AOB  = (bf16*)(ws + 159514624);           // 33,587,200
  float* RES1 = (float*)(ws + 193101824);         // 67,174,400  (total 260,276,224)
  bf16* H1B  = QKVB;  // alias: FC1 output (134,348,800 B) reuses dead QKV+attn buffers

  float* out = (float*)d_out;

  // 1) weights -> bf16
  cvt_kernel<<<2048, 256, 0, stream>>>(qkv_w, WQKV, 3 * 1024 * 1024);
  cvt_kernel<<<2048, 256, 0, stream>>>(proj_w, WPROJ, 1024 * 1024);
  cvt_kernel<<<2048, 256, 0, stream>>>(fc1_w, WFC1, 4096 * 1024);
  cvt_kernel<<<2048, 256, 0, stream>>>(fc2_w, WFC2, 4096 * 1024);
  // 2) LN1
  ln_kernel<<<T, 256, 0, stream>>>(hidden, n1g, n1b, XBF);
  // 3) QKV GEMM  [T,1024] x [3072,1024]^T -> bf16 [T,3072]
  gemm_kernel<0><<<dim3(3072 / 128, (T + 127) / 128), 256, 0, stream>>>(
      XBF, WQKV, qkv_b, nullptr, nullptr, QKVB, nullptr, T, 1024, 3072);
  // 4) attention -> bf16 [T,1024]
  attn_kernel<<<dim3(16 * 16, (1025 + 63) / 64), 128, 0, stream>>>(QKVB, AOB);
  // 5) proj GEMM + ls1 residual -> f32 RES1
  gemm_kernel<1><<<dim3(1024 / 128, (T + 127) / 128), 256, 0, stream>>>(
      AOB, WPROJ, proj_b, hidden, ls1, nullptr, RES1, T, 1024, 1024);
  // 6) LN2
  ln_kernel<<<T, 256, 0, stream>>>(RES1, n2g, n2b, XBF);
  // 7) FC1 + GELU -> bf16 [T,4096]
  gemm_kernel<2><<<dim3(4096 / 128, (T + 127) / 128), 256, 0, stream>>>(
      XBF, WFC1, fc1_b, nullptr, nullptr, H1B, nullptr, T, 1024, 4096);
  // 8) FC2 + ls2 residual -> f32 d_out
  gemm_kernel<1><<<dim3(1024 / 128, (T + 127) / 128), 256, 0, stream>>>(
      H1B, WFC2, fc2_b, RES1, ls2, nullptr, out, T, 4096, 1024);
}
